// SDOT_11579231830512
// MI455X (gfx1250) — matrix operand order repaired and checked by reading, not compiled, not run on hardware
//
#include <hip/hip_runtime.h>
#include <hip/hip_fp16.h>

typedef __attribute__((ext_vector_type(16))) _Float16 v16h;
typedef __attribute__((ext_vector_type(8)))  float    v8f;

#define EPS   0.01f
#define L2E   1.4426950408889634f
#define LN2   0.6931471805599453f
#define N_SRC 32768
#define M_TGT 8192
#define DD    64

// ---------------------------------------------------------------------------
// Prologue 1: per-row sum of squares (+ optional f32->f16 row conversion).
// One wave (32 lanes) per row of 64 floats; deterministic xor-tree reduction.
// ---------------------------------------------------------------------------
__global__ void rowprep_kernel(const float* __restrict__ src,
                               _Float16* __restrict__ dsth,
                               float* __restrict__ sums, int rows) {
  const int wave = threadIdx.x >> 5;
  const int lane = threadIdx.x & 31;
  const int row  = blockIdx.x * 8 + wave;
  if (row >= rows) return;
  const float* r = src + (size_t)row * DD;
  float v0 = r[lane];
  float v1 = r[lane + 32];
  if (dsth) {
    dsth[(size_t)row * DD + lane]      = (_Float16)v0;
    dsth[(size_t)row * DD + lane + 32] = (_Float16)v1;
  }
  float ss = v0 * v0 + v1 * v1;
  #pragma unroll
  for (int off = 16; off; off >>= 1) ss += __shfl_xor(ss, off, 32);
  if (lane == 0) sums[row] = ss;
}

// ---------------------------------------------------------------------------
// Prologue 2 (single block, deterministic): mean ||y||^2, mean psi, and
// q[m] = a*psi[m] - a*inv_mv*||y_m||^2   with a = log2(e)/eps.
// ---------------------------------------------------------------------------
__global__ void stats_kernel(const float* __restrict__ sy,
                             const float* __restrict__ psi,
                             float* __restrict__ q,
                             float* __restrict__ consts) {
  __shared__ float red[1024];
  __shared__ float sh_inv_mv;
  const int t = threadIdx.x;

  float ls = 0.f, lp = 0.f;
  for (int i = t; i < M_TGT; i += 1024) { ls += sy[i]; lp += psi[i]; }

  red[t] = ls; __syncthreads();
  #pragma unroll
  for (int off = 512; off; off >>= 1) {
    if (t < off) red[t] += red[t + off];
    __syncthreads();
  }
  const float mv = red[0] / (float)M_TGT;
  __syncthreads();

  red[t] = lp; __syncthreads();
  #pragma unroll
  for (int off = 512; off; off >>= 1) {
    if (t < off) red[t] += red[t + off];
    __syncthreads();
  }
  const float mean_psi = red[0] / (float)M_TGT;

  if (t == 0) {
    consts[0] = 1.f / mv;
    consts[1] = mean_psi;
    sh_inv_mv = 1.f / mv;
  }
  __syncthreads();

  const float a  = L2E / EPS;
  const float ai = a * sh_inv_mv;
  for (int i = t; i < M_TGT; i += 1024) q[i] = a * psi[i] - ai * sy[i];
}

// ---------------------------------------------------------------------------
// Main fused GEMM + online logsumexp kernel.
// Block = 256 threads (8 waves) handles 128 source rows. Each wave owns a
// 16-row A fragment (K=64 -> two v16h fragments) kept in VGPRs for the whole
// M sweep. Y (f16) + q stream through double-buffered LDS in 64-column
// chunks. All four 16x16 accumulator tiles of a chunk are materialized
// before a single batched logsumexp update (one rescale exp per row per
// chunk -> 1.25 exp2/element instead of 2).
// ---------------------------------------------------------------------------
__global__ void __launch_bounds__(256, 1)
sinkhorn_kernel(const float* __restrict__ X, const _Float16* __restrict__ Yh,
                const float* __restrict__ q, const float* __restrict__ sx,
                const float* __restrict__ consts, float* __restrict__ out) {
  __shared__ __align__(32) _Float16 ybuf[2][64 * DD];   // 2 x 8 KB
  __shared__ float qbuf[2][64];                         // 2 x 256 B

  const int t    = threadIdx.x;
  const int wave = t >> 5;
  const int lane = t & 31;
  const int l15  = lane & 15;
  const int ksel = lane >> 4;                 // which K-half this lane holds
  const int n0   = blockIdx.x * 128 + wave * 16;

  const float inv_mv   = consts[0];
  const float mean_psi = consts[1];
  const float a        = L2E / EPS;
  const float cmul     = 2.f * a * inv_mv;    // scale on the dot product

  // ---- Build A fragments (convert this wave's 16 X rows to f16 once) ----
  // A 16x32 f16 layout: lane holds row n0+l15; halves 0..15 = K base+0..15,
  // base = kc*32 + ksel*16.
  v16h a0, a1;
  {
    const float* xr = X + (size_t)(n0 + l15) * DD;
    #pragma unroll
    for (int kc = 0; kc < 2; ++kc) {
      const int kb = kc * 32 + ksel * 16;
      v16h av;
      #pragma unroll
      for (int i = 0; i < 8; ++i) av[i]     = (_Float16)xr[kb + i];
      #pragma unroll
      for (int i = 0; i < 8; ++i) av[8 + i] = (_Float16)xr[kb + 8 + i];
      if (kc == 0) a0 = av; else a1 = av;
    }
  }

  // Online logsumexp state: lane holds 8 rows (row = v + 8*ksel), one column.
  float mx[8], sm[8];
  #pragma unroll
  for (int v = 0; v < 8; ++v) { mx[v] = -3.0e38f; sm[v] = 0.f; }

  const uint4* gy = (const uint4*)Yh;         // Yh is row-major M x 64 halves
  uint4 g0 = gy[t * 2];
  uint4 g1 = gy[t * 2 + 1];
  float qg = (t < 64) ? q[t] : 0.f;

  const int NCH = M_TGT / 64;                 // 128 chunks of 64 columns
  for (int c = 0; c < NCH; ++c) {
    uint4* lb = (uint4*)&ybuf[c & 1][0];
    lb[t * 2]     = g0;                       // cooperative 8 KB stage
    lb[t * 2 + 1] = g1;
    if (t < 64) qbuf[c & 1][t] = qg;
    __syncthreads();                          // single barrier per chunk
    if (c + 1 < NCH) {                        // prefetch next chunk
      g0 = gy[(size_t)(c + 1) * 512 + t * 2];
      g1 = gy[(size_t)(c + 1) * 512 + t * 2 + 1];
      if (t < 64) qg = q[(c + 1) * 64 + t];
    }
    const _Float16* yb = &ybuf[c & 1][0];

    // ---- 4 WMMA subtiles of 16 columns; keep all accumulators live ----
    v8f dt0, dt1, dt2, dt3;
    #define SUBTILE(i, DT)                                                   \
    {                                                                        \
      const _Float16* br = yb + ((i) * 16 + l15) * DD + ksel * 16;           \
      v16h b0 = *(const v16h*)(br);                                          \
      v16h b1 = *(const v16h*)(br + 32);                                     \
      v8f dd = {};                                                           \
      dd = __builtin_amdgcn_wmma_f32_16x16x32_f16(false, a0, false, b0,      \
                                                  (short)0, dd, false, false);\
      DT = __builtin_amdgcn_wmma_f32_16x16x32_f16(false, a1, false, b1,      \
                                                  (short)0, dd, false, false);\
    }
    SUBTILE(0, dt0)
    SUBTILE(1, dt1)
    SUBTILE(2, dt2)
    SUBTILE(3, dt3)
    #undef SUBTILE

    const float qv0 = qbuf[c & 1][0 * 16 + l15];
    const float qv1 = qbuf[c & 1][1 * 16 + l15];
    const float qv2 = qbuf[c & 1][2 * 16 + l15];
    const float qv3 = qbuf[c & 1][3 * 16 + l15];

    // ---- Batched online update: one rescale per row per chunk ----
    #pragma unroll
    for (int v = 0; v < 8; ++v) {
      float z0 = fmaf(cmul, dt0[v], qv0);
      float z1 = fmaf(cmul, dt1[v], qv1);
      float z2 = fmaf(cmul, dt2[v], qv2);
      float z3 = fmaf(cmul, dt3[v], qv3);
      float tm = fmaxf(fmaxf(z0, z1), fmaxf(z2, z3));
      float nm = fmaxf(mx[v], tm);
      float s  = sm[v] * __builtin_amdgcn_exp2f(mx[v] - nm);
      s += __builtin_amdgcn_exp2f(z0 - nm);
      s += __builtin_amdgcn_exp2f(z1 - nm);
      s += __builtin_amdgcn_exp2f(z2 - nm);
      s += __builtin_amdgcn_exp2f(z3 - nm);
      sm[v] = s;
      mx[v] = nm;
    }
  }

  // ---- Merge the 16 lanes that share each row (xor butterfly, width 16) ----
  #pragma unroll
  for (int v = 0; v < 8; ++v) {
    float m = mx[v], s = sm[v];
    #pragma unroll
    for (int off = 8; off; off >>= 1) {
      float om = __shfl_xor(m, off, 32);
      float os = __shfl_xor(s, off, 32);
      float nm = fmaxf(m, om);
      s = s * __builtin_amdgcn_exp2f(m - nm)
        + os * __builtin_amdgcn_exp2f(om - nm);
      m = nm;
    }
    mx[v] = m; sm[v] = s;
  }

  if (l15 == 0) {
    const float oscale = -EPS * LN2;          // -eps*ln2
    const float pm     = a * inv_mv;          // row-term coefficient
    const int   base   = n0 + ksel * 8;
    #pragma unroll
    for (int v = 0; v < 8; ++v) {
      const int n = base + v;
      float val = oscale * (mx[v] + __builtin_amdgcn_logf(sm[v]) - pm * sx[n])
                + mean_psi;
      out[n] = val;
    }
  }
}

// ---------------------------------------------------------------------------
extern "C" void kernel_launch(void* const* d_in, const int* in_sizes, int n_in,
                              void* d_out, int out_size, void* d_ws,
                              size_t ws_size, hipStream_t stream) {
  (void)in_sizes; (void)n_in; (void)out_size; (void)ws_size;
  const float* X   = (const float*)d_in[0];   // [N, 64]
  const float* Y   = (const float*)d_in[1];   // [M, 64]
  const float* psi = (const float*)d_in[2];   // [M]
  float* out = (float*)d_out;                 // [N]

  char* ws = (char*)d_ws;
  _Float16* Yh     = (_Float16*)ws;                       // 1 MB
  float*    sy     = (float*)(ws + (1u << 20));           // 32 KB
  float*    sx     = (float*)(ws + (1u << 20) + 32768);   // 128 KB
  float*    q      = (float*)(ws + (1u << 20) + 32768 + 131072);
  float*    consts = (float*)(ws + (1u << 20) + 32768 + 131072 + 32768);

  rowprep_kernel<<<M_TGT / 8, 256, 0, stream>>>(Y, Yh, sy, M_TGT);
  rowprep_kernel<<<N_SRC / 8, 256, 0, stream>>>(X, nullptr, sx, N_SRC);
  stats_kernel<<<1, 1024, 0, stream>>>(sy, psi, q, consts);
  sinkhorn_kernel<<<N_SRC / 128, 256, 0, stream>>>(X, Yh, q, sx, consts, out);
}
